// deformable_self_attention_46024869544005
// MI455X (gfx1250) — matrix operand order repaired and checked by reading, not compiled, not run on hardware
//
#include <hip/hip_runtime.h>
#include <hip/hip_bf16.h>

#define DM    256          // d_model
#define NSP   8            // sampling points
#define BATCH 2
#define NQ    10000
#define MROWS (BATCH * NQ) // 20000 rows, divisible by 16
#define VD    16
#define VH    64
#define VW    64
#define SPAT  (VD * VH * VW)   // 65536 voxels per batch

typedef __attribute__((ext_vector_type(16))) __bf16 v16bf;
typedef __attribute__((ext_vector_type(8)))  __bf16 v8bf;
typedef __attribute__((ext_vector_type(8)))  float  v8f;

// ---------------- workspace layout (bytes) ----------------
static constexpr size_t OFF_VOLT = 0;                               // bf16 vol (B,D,H,W,C): 67108864
static constexpr size_t OFF_AGG  = OFF_VOLT + (size_t)BATCH*SPAT*DM*2;   // bf16 agg (20000,256): 10240000
static constexpr size_t OFF_HID  = OFF_AGG  + (size_t)MROWS*DM*2;        // bf16 hidden (20000,256)
static constexpr size_t OFF_WC   = OFF_HID  + (size_t)MROWS*DM*2;        // bf16 Wc (256,256)
static constexpr size_t OFF_WOB  = OFF_WC   + (size_t)DM*DM*2;           // bf16 Wo (256,256)
static constexpr size_t OFF_BC   = OFF_WOB  + (size_t)DM*DM*2;           // f32 bc (256)

// ---------------- kernel A: fuse Wc = Ww@Wv, bc = Ww@bv + bw; Wo -> bf16 ----------------
__global__ __launch_bounds__(256)
void prep_weights_k(const float* __restrict__ Wv, const float* __restrict__ bv,
                    const float* __restrict__ Ww, const float* __restrict__ bw,
                    const float* __restrict__ Wo,
                    __bf16* __restrict__ WcB, float* __restrict__ bc,
                    __bf16* __restrict__ WoB) {
  const int n = blockIdx.x;       // output row
  const int k = threadIdx.x;      // output col
  float s = 0.f;
  #pragma unroll 4
  for (int j = 0; j < DM; ++j)
    s += Ww[n * DM + j] * Wv[j * DM + k];   // Ww row is block-uniform (scalarizes)
  WcB[n * DM + k] = (__bf16)s;
  WoB[n * DM + k] = (__bf16)Wo[n * DM + k];
  if (k == 0) {
    float t = bw[n];
    for (int j = 0; j < DM; ++j) t += Ww[n * DM + j] * bv[j];
    bc[n] = t;
  }
}

// ---------------- kernel T: (B,C,S) f32 -> (B,S,C) bf16, LDS tiled transpose ----------------
__global__ __launch_bounds__(1024)
void transpose_vol_k(const float* __restrict__ vin, __bf16* __restrict__ vout) {
  __shared__ float tile[32][33];
  const int b  = blockIdx.z;
  const int c0 = blockIdx.y * 32;
  const int s0 = blockIdx.x * 32;
  // coalesced read along spatial dim
  tile[threadIdx.y][threadIdx.x] =
      vin[((size_t)b * DM + (c0 + threadIdx.y)) * SPAT + (s0 + threadIdx.x)];
  __syncthreads();
  // coalesced write along channel dim
  vout[((size_t)b * SPAT + (s0 + threadIdx.y)) * DM + (c0 + threadIdx.x)] =
      (__bf16)tile[threadIdx.x][threadIdx.y];
}

// ---------------- kernel B: per-query offsets + softmax + trilinear gather + attn-weighted sum ----------------
// One wave32 per query. Lanes 0..23 compute delta_p (8 pts x 3), lanes 24..31 attn logits.
// Then all lanes gather 8 channels each (256 = 32*8) from the bf16 channel-contiguous volume.
__global__ __launch_bounds__(256)
void sample_agg_k(const float* __restrict__ query, const float* __restrict__ ref,
                  const __bf16* __restrict__ volT,
                  const float* __restrict__ Wdp, const float* __restrict__ bdp,
                  const float* __restrict__ Wa,  const float* __restrict__ ba,
                  __bf16* __restrict__ agg) {
  __shared__ float qs[8][DM];
  const int wave = threadIdx.x >> 5;
  const int lane = threadIdx.x & 31;
  const int q    = blockIdx.x * 8 + wave;          // 2500*8 = 20000 exactly
  const int b    = q / NQ;

  // stage query vector in LDS (each lane brings 8 floats)
  const float4* qp = (const float4*)(query + (size_t)q * DM);
  float4 qa = qp[lane * 2 + 0];
  float4 qb = qp[lane * 2 + 1];
  *(float4*)&qs[wave][lane * 8 + 0] = qa;
  *(float4*)&qs[wave][lane * 8 + 4] = qb;
  __syncthreads();

  // lane -> one fused-linear output (24 offsets + 8 attn logits = 32)
  const float* wrow = (lane < 24) ? (Wdp + (size_t)lane * DM)
                                  : (Wa  + (size_t)(lane - 24) * DM);
  float acc = (lane < 24) ? bdp[lane] : ba[lane - 24];
  #pragma unroll 8
  for (int k = 0; k < DM; ++k) acc += qs[wave][k] * wrow[k];

  // softmax over the 8 logits (lanes 24..31), replicated to every lane
  float lg[NSP];
  #pragma unroll
  for (int j = 0; j < NSP; ++j) lg[j] = __shfl(acc, 24 + j, 32);
  float mx = lg[0];
  #pragma unroll
  for (int j = 1; j < NSP; ++j) mx = fmaxf(mx, lg[j]);
  float ssum = 0.f;
  #pragma unroll
  for (int j = 0; j < NSP; ++j) { lg[j] = __expf(lg[j] - mx); ssum += lg[j]; }
  const float inv = 1.f / ssum;

  const float* rp = ref + (size_t)q * 3;
  const float r0 = rp[0], r1 = rp[1], r2 = rp[2];

  float facc[8];
  #pragma unroll
  for (int j = 0; j < 8; ++j) facc[j] = 0.f;

  const size_t vbase = (size_t)b * SPAT * DM + (size_t)lane * 8;

  for (int p = 0; p < NSP; ++p) {
    const float d0 = __shfl(acc, 3 * p + 0, 32);
    const float d1 = __shfl(acc, 3 * p + 1, 32);
    const float d2 = __shfl(acc, 3 * p + 2, 32);
    // grid = p_delta_p[..., (1,0,2)] ; grid_sample: x->W, y->H, z->D
    const float gx = r1 + d1, gy = r0 + d0, gz = r2 + d2;
    const float ix = (gx + 1.f) * 0.5f * (float)(VW - 1);
    const float iy = (gy + 1.f) * 0.5f * (float)(VH - 1);
    const float iz = (gz + 1.f) * 0.5f * (float)(VD - 1);
    const float fx0 = floorf(ix), fy0 = floorf(iy), fz0 = floorf(iz);
    const int   x0 = (int)fx0,   y0 = (int)fy0,   z0 = (int)fz0;
    const float fx = ix - fx0, fy = iy - fy0, fz = iz - fz0;
    const float ap = lg[p] * inv;   // fold attention weight into corner weight

    #pragma unroll
    for (int dz = 0; dz < 2; ++dz)
      #pragma unroll
      for (int dy = 0; dy < 2; ++dy)
        #pragma unroll
        for (int dx = 0; dx < 2; ++dx) {
          const int xi = x0 + dx, yi = y0 + dy, zi = z0 + dz;
          const bool valid = (xi >= 0) & (xi < VW) & (yi >= 0) & (yi < VH) &
                             (zi >= 0) & (zi < VD);
          if (!valid) continue;     // zero padding -> skip (wave-uniform branch)
          const float w = ap * (dx ? fx : 1.f - fx) * (dy ? fy : 1.f - fy)
                             * (dz ? fz : 1.f - fz);
          const v8bf v = *(const v8bf*)(volT + vbase +
                            (size_t)((zi * VH + yi) * VW + xi) * DM);
          #pragma unroll
          for (int j = 0; j < 8; ++j) facc[j] += w * (float)v[j];
        }
  }

  v8bf o;
  #pragma unroll
  for (int j = 0; j < 8; ++j) o[j] = (__bf16)facc[j];
  *(v8bf*)(agg + (size_t)q * DM + lane * 8) = o;
}

// ---------------- GEMM: out[M,256] = A[M,256] @ W[256,256]^T + bias ; optional mask ----------------
// One wave computes two 16x16 tiles, each via 8x v_wmma_f32_16x16x32_bf16 (K=256).
__global__ __launch_bounds__(256)
void gemm256_k(const __bf16* __restrict__ A, const __bf16* __restrict__ W,
               const float* __restrict__ bias, void* __restrict__ out,
               int outIsBf16, const unsigned char* __restrict__ maskPtr) {
  const int wave  = threadIdx.x >> 5;
  const int lane  = threadIdx.x & 31;
  const int half  = lane >> 4;          // K-half selector
  const int l16   = lane & 15;
  const int tileM = blockIdx.x * 16;

  const __bf16* arow = A + (size_t)(tileM + l16) * DM;

  for (int t = 0; t < 2; ++t) {
    const int tileN = (wave * 2 + t) * 16;
    const __bf16* brow = W + (size_t)(tileN + l16) * DM;
    v8f c = {};
    #pragma unroll
    for (int ks = 0; ks < DM; ks += 32) {
      // A fragment: lane(half,row) holds 16 contiguous K at ks+16h
      v16bf a = *(const v16bf*)(arow + ks + 16 * half);
      // B fragment: lane holds col tileN+l16, 16 contiguous K at ks+16h
      v16bf bm = *(const v16bf*)(brow + ks + 16 * half);
      c = __builtin_amdgcn_wmma_f32_16x16x32_bf16(false, a, false, bm,
                                                  (short)0, c, false, false);
    }
    const int n = tileN + l16;
    const float bn = bias[n];
    #pragma unroll
    for (int r = 0; r < 8; ++r) {
      const int m = tileM + r + 8 * half;     // D layout: VGPR r -> M = r + 8*half
      float v = c[r] + bn;
      if (maskPtr) {
        const int bb = m / NQ, nn = m % NQ;
        const unsigned char mm = maskPtr[bb * 2 * NQ + nn] |
                                 maskPtr[bb * 2 * NQ + NQ + nn];
        v = mm ? v : 0.f;
      }
      if (outIsBf16) ((__bf16*)out)[(size_t)m * DM + n] = (__bf16)v;
      else           ((float*)out)[(size_t)m * DM + n]  = v;
    }
  }
}

// ---------------- host launcher ----------------
extern "C" void kernel_launch(void* const* d_in, const int* in_sizes, int n_in,
                              void* d_out, int out_size, void* d_ws, size_t ws_size,
                              hipStream_t stream) {
  const float* query = (const float*)d_in[0];
  const float* refp  = (const float*)d_in[1];
  const float* imgf  = (const float*)d_in[2];
  const unsigned char* mask = (const unsigned char*)d_in[3];
  const float* Wdp = (const float*)d_in[4];
  const float* bdp = (const float*)d_in[5];
  const float* Wa  = (const float*)d_in[6];
  const float* ba  = (const float*)d_in[7];
  const float* Wv  = (const float*)d_in[8];
  const float* bv  = (const float*)d_in[9];
  const float* Ww  = (const float*)d_in[10];
  const float* bw  = (const float*)d_in[11];
  const float* Wo  = (const float*)d_in[12];
  const float* bo  = (const float*)d_in[13];

  char* ws = (char*)d_ws;
  __bf16* volT = (__bf16*)(ws + OFF_VOLT);
  __bf16* agg  = (__bf16*)(ws + OFF_AGG);
  __bf16* hid  = (__bf16*)(ws + OFF_HID);
  __bf16* WcB  = (__bf16*)(ws + OFF_WC);
  __bf16* WoB  = (__bf16*)(ws + OFF_WOB);
  float*  bc   = (float*)(ws + OFF_BC);

  // 1) fuse value/W projections: Wc = Ww@Wv, bc = Ww@bv + bw ; Wo -> bf16
  prep_weights_k<<<DM, DM, 0, stream>>>(Wv, bv, Ww, bw, Wo, WcB, bc, WoB);
  // 2) volume (B,C,D,H,W) f32 -> (B,D,H,W,C) bf16 (channel-contiguous gathers)
  transpose_vol_k<<<dim3(SPAT / 32, DM / 32, BATCH), dim3(32, 32), 0, stream>>>(imgf, volT);
  // 3) offsets + softmax + trilinear sampling + attention-weighted reduction
  sample_agg_k<<<MROWS / 8, 256, 0, stream>>>(query, refp, volT, Wdp, bdp, Wa, ba, agg);
  // 4) hidden = mask * (agg @ Wc^T + bc)   [WMMA, bf16 out]
  gemm256_k<<<MROWS / 16, 256, 0, stream>>>(agg, WcB, bc, hid, 1, mask);
  // 5) out = hidden @ Wo^T + bo            [WMMA, f32 out]
  gemm256_k<<<MROWS / 16, 256, 0, stream>>>(hid, WoB, bo, d_out, 0, nullptr);
}
